// PWCDCNet_23201413333507
// MI455X (gfx1250) — matrix operand order repaired and checked by reading, not compiled, not run on hardware
//
#include <hip/hip_runtime.h>
#include <cstdio>
#include <cstring>

typedef _Float16 f16;
typedef __attribute__((ext_vector_type(16))) _Float16 v16h;
typedef __attribute__((ext_vector_type(8)))  float    v8f;
typedef __attribute__((ext_vector_type(4)))  unsigned u32x4;

#define LEAK 0.1f

union Frag16 { v16h v; f16 h[16]; unsigned u[8]; u32x4 q[2]; };

__device__ __forceinline__ int imin(int a, int b) { return a < b ? a : b; }

// ---------------------------------------------------------------------------
// Implicit-GEMM conv via V_WMMA_F32_16X16X32_F16.
// Wave tile: 16 pixels (M) x 64 output channels (4 x 16 N sub-tiles).
// Weights pre-packed [K*K][CinP][CoutP] (f16, zero-padded) so that a lane's
// B fragment (lane%16 = N, halves = K rows) is 16 f16 loads at stride CoutP.
// Activations are f16 NHWC views with channel-stride % 8 == 0 and read-offset
// % 8 == 0, so a lane's A fragment is two aligned b128 loads
// (K = base..base+7 and base+16..base+23 per the ISA 16-bit A layout).
// No LDS, no barriers. f32 accumulation, fused bias + leaky ReLU.
// ---------------------------------------------------------------------------
__global__ __launch_bounds__(256) void conv_wmma_kernel(
    const f16* __restrict__ in, int in_cs, int in_coff, int CinP,
    f16* __restrict__ out, int out_cs, int out_coff, int Cout, int CoutP,
    const f16* __restrict__ wt, const float* __restrict__ bias,
    int H, int Wd, int OH, int OW,
    int K, int stride, int dil, int pad_h, int pad_w, int leaky)
{
    const int lane = threadIdx.x & 31;
    const int wv   = threadIdx.x >> 5;
    const int b    = blockIdx.z;
    const int n0   = blockIdx.y << 6;          // 64 couts per block
    const int npix = OH * OW;
    const int m0   = (blockIdx.x << 7) + (wv << 4);
    const int lh   = lane >> 4;                // lane/16
    const int ml   = lane & 15;                // lane%16
    const int nsub = imin(4, (CoutP - n0) >> 4);

    v8f z8 = {0.f, 0.f, 0.f, 0.f, 0.f, 0.f, 0.f, 0.f};
    v8f acc[4] = {z8, z8, z8, z8};

    const f16* inb = in + (long)b * H * Wd * in_cs + in_coff;

    // per-lane A-matrix row -> one output pixel (A layout: M = lane%16)
    const int  pA = m0 + ml;
    const bool pv = pA < npix;
    const int  oy = pv ? pA / OW : 0;
    const int  ox = pv ? pA % OW : 0;
    const int  abase = lh << 3;                // K half-select per lane

    for (int kh = 0; kh < K; ++kh) {
      int  iy  = oy * stride - pad_h + kh * dil;
      bool yok = pv && iy >= 0 && iy < H;
      for (int kw = 0; kw < K; ++kw) {
        int  ix = ox * stride - pad_w + kw * dil;
        bool ok = yok && ix >= 0 && ix < Wd;
        const f16* pin = inb + ((long)iy * Wd + ix) * in_cs;
        const f16* wkk = wt + (long)(kh * K + kw) * CinP * CoutP;

        for (int c0 = 0; c0 < CinP; c0 += 32) {
          Frag16 af;
          if (ok) {
            af.q[0] = *(const u32x4*)(pin + c0 + abase);        // K base..base+7
            af.q[1] = *(const u32x4*)(pin + c0 + abase + 16);   // K base+16..+23
          } else {
            u32x4 zq = {0u, 0u, 0u, 0u};
            af.q[0] = zq; af.q[1] = zq;
          }

          const f16* wrow = wkk + (long)(c0 + abase) * CoutP + n0 + ml;
          #pragma unroll
          for (int s = 0; s < 4; ++s) {
            if (s < nsub) {
              Frag16 bf;
              #pragma unroll
              for (int e = 0; e < 16; ++e)
                bf.h[e] = wrow[(long)(e < 8 ? e : e + 8) * CoutP + (s << 4)];
              acc[s] = __builtin_amdgcn_wmma_f32_16x16x32_f16(
                  false, af.v, false, bf.v, (short)0, acc[s], false, false);
            }
          }
        }
      }
    }

    // epilogue: C/D layout N = lane%16, M = vgpr + 8*(lane/16)
    #pragma unroll
    for (int s = 0; s < 4; ++s) {
      if (s < nsub) {
        int n = n0 + (s << 4) + ml;
        if (n < Cout) {
          float bv = bias[n];
          #pragma unroll
          for (int v = 0; v < 8; ++v) {
            int m = m0 + v + (lh << 3);
            if (m < npix) {
              float val = acc[s][v] + bv;
              if (leaky) val = val > 0.f ? val : LEAK * val;
              out[((long)b * npix + m) * out_cs + out_coff + n] = (f16)val;
            }
          }
        }
      }
    }
}

// ---------------------------------------------------------------------------
// Support kernels
// ---------------------------------------------------------------------------
__global__ void fill0_kernel(u32x4* __restrict__ p, long n16) {
  long i = (long)blockIdx.x * blockDim.x + threadIdx.x;
  u32x4 z = {0u, 0u, 0u, 0u};
  if (i < n16) p[i] = z;
}

// f32 [KK][Cin][Cout] -> f16 [KK][CinP][CoutP], zero padded
__global__ void pack_w_kernel(const float* __restrict__ w, f16* __restrict__ o,
                              int KK, int Cin, int Cout, int CinP, int CoutP) {
  long total = (long)KK * CinP * CoutP;
  long i = (long)blockIdx.x * blockDim.x + threadIdx.x;
  if (i >= total) return;
  int n = (int)(i % CoutP); long t = i / CoutP;
  int c = (int)(t % CinP);  int kk = (int)(t / CinP);
  f16 v = (f16)0.f;
  if (c < Cin && n < Cout) v = (f16)w[((long)kk * Cin + c) * Cout + n];
  o[i] = v;
}

__global__ void split_kernel(const float* __restrict__ in, f16* __restrict__ im1,
                             f16* __restrict__ im2, long npix) {
  long i = (long)blockIdx.x * blockDim.x + threadIdx.x;
  if (i >= npix * 3) return;
  long p = i / 3; int c = (int)(i % 3);
  im1[p * 8 + c] = (f16)in[p * 6 + c];       // im stride padded to 8
  im2[p * 8 + c] = (f16)in[p * 6 + 3 + c];
}

__global__ void copy_ch_kernel(const f16* __restrict__ s, int s_cs, int s_coff,
                               f16* __restrict__ d, int d_cs, int d_coff, int C, long np) {
  long i = (long)blockIdx.x * blockDim.x + threadIdx.x;
  if (i >= np * C) return;
  long p = i / C; int c = (int)(i % C);
  d[p * d_cs + d_coff + c] = s[p * s_cs + s_coff + c];
}

__global__ void cost_volume_kernel(const f16* __restrict__ c1, int c1_cs,
                                   const f16* __restrict__ wp, int wp_cs, int C,
                                   f16* __restrict__ out, int o_cs, int o_coff,
                                   int N, int H, int W) {
  long total = (long)N * H * W * 81;
  long i = (long)blockIdx.x * blockDim.x + threadIdx.x;
  if (i >= total) return;
  int o = (int)(i % 81); long pix = i / 81;
  int x = (int)(pix % W); int y = (int)((pix / W) % H); int b = (int)(pix / ((long)W * H));
  int dy = o / 9 - 4, dx = o % 9 - 4;
  int yy = y + dy, xx = x + dx;
  float s = 0.f;
  if (yy >= 0 && yy < H && xx >= 0 && xx < W) {
    const f16* p1 = c1 + ((long)b * H * W + (long)y  * W + x ) * c1_cs;
    const f16* p2 = wp + ((long)b * H * W + (long)yy * W + xx) * wp_cs;
    for (int c = 0; c < C; ++c) s += (float)p1[c] * (float)p2[c];
    s *= (1.f / (float)C);
  }
  s = s > 0.f ? s : LEAK * s;
  out[((long)b * H * W + (long)y * W + x) * o_cs + o_coff + o] = (f16)s;
}

__global__ void warp_kernel(const f16* __restrict__ im, int im_cs, int C,
                            const f16* __restrict__ fb, int f_cs, int f_coff, float scale,
                            f16* __restrict__ out, int N, int H, int W) {
  long total = (long)N * H * W * C;
  long i = (long)blockIdx.x * blockDim.x + threadIdx.x;
  if (i >= total) return;
  int c = (int)(i % C); long pix = i / C;
  int x = (int)(pix % W); int y = (int)((pix / W) % H); int b = (int)(pix / ((long)W * H));
  long pb = (long)b * H * W;
  const f16* fp = fb + (pb + (long)y * W + x) * f_cs + f_coff;
  float xs = (float)x + (float)fp[0] * scale;
  float ys = (float)y + (float)fp[1] * scale;
  float x0 = floorf(xs), y0 = floorf(ys);
  float fx = xs - x0, fy = ys - y0;
  int x0i = (int)fminf(fmaxf(x0,       0.f), (float)(W - 1));
  int x1i = (int)fminf(fmaxf(x0 + 1.f, 0.f), (float)(W - 1));
  int y0i = (int)fminf(fmaxf(y0,       0.f), (float)(H - 1));
  int y1i = (int)fminf(fmaxf(y0 + 1.f, 0.f), (float)(H - 1));
  float Ia = (float)im[(pb + (long)y0i * W + x0i) * im_cs + c];
  float Ib = (float)im[(pb + (long)y1i * W + x0i) * im_cs + c];
  float Ic = (float)im[(pb + (long)y0i * W + x1i) * im_cs + c];
  float Id = (float)im[(pb + (long)y1i * W + x1i) * im_cs + c];
  float v = (1.f - fx) * (1.f - fy) * Ia + (1.f - fx) * fy * Ib
          + fx * (1.f - fy) * Ic + fx * fy * Id;
  out[(pb + (long)y * W + x) * C + c] = (f16)v;
}

// 4x4 stride-2 SAME conv_transpose (cout == 2 everywhere -> scalar VALU).
// Weights are the packed [16][CinP][CoutP] tensors.
__global__ void deconv_kernel(const f16* __restrict__ in, int in_cs, int in_coff, int Cin,
                              const f16* __restrict__ wt, int CinP, int CoutP,
                              const float* __restrict__ bias, int Cout,
                              f16* __restrict__ out, int o_cs, int o_coff,
                              int N, int IH, int IW, int OH, int OW) {
  long total = (long)N * OH * OW * Cout;
  long i = (long)blockIdx.x * blockDim.x + threadIdx.x;
  if (i >= total) return;
  int co = (int)(i % Cout); long pix = i / Cout;
  int xo = (int)(pix % OW); int yo = (int)((pix / OW) % OH); int b = (int)(pix / ((long)OW * OH));
  float s = bias[co];
  for (int kh = 0; kh < 4; ++kh) {
    int yt = yo + 1 - kh;
    if (yt & 1) continue;
    int yi = yt >> 1;
    if (yi < 0 || yi >= IH) continue;
    for (int kw = 0; kw < 4; ++kw) {
      int xt = xo + 1 - kw;
      if (xt & 1) continue;
      int xi = xt >> 1;
      if (xi < 0 || xi >= IW) continue;
      const f16* pin = in + ((long)b * IH * IW + (long)yi * IW + xi) * in_cs + in_coff;
      const f16* pw  = wt + (long)(kh * 4 + kw) * CinP * CoutP + co;
      for (int ci = 0; ci < Cin; ++ci)
        s += (float)pin[ci] * (float)pw[(long)ci * CoutP];
    }
  }
  out[((long)b * OH * OW + (long)yo * OW + xo) * o_cs + o_coff + co] = (f16)s;
}

__global__ void add_flow_kernel(const f16* __restrict__ a, const f16* __restrict__ bb,
                                float* __restrict__ o, long n) {
  long i = (long)blockIdx.x * blockDim.x + threadIdx.x;
  if (i < n) o[i] = (float)a[i] + (float)bb[i];
}

// ---------------------------------------------------------------------------
// Host orchestration
// ---------------------------------------------------------------------------
struct SpecT { const char* n; int k, cin, cout; };
static const SpecT SPECS[63] = {
 {"conv1a",3,3,16},{"conv1aa",3,16,16},{"conv1b",3,16,16},
 {"conv2a",3,16,32},{"conv2aa",3,32,32},{"conv2b",3,32,32},
 {"conv3a",3,32,64},{"conv3aa",3,64,64},{"conv3b",3,64,64},
 {"conv4a",3,64,96},{"conv4aa",3,96,96},{"conv4b",3,96,96},
 {"conv5a",3,96,128},{"conv5aa",3,128,128},{"conv5b",3,128,128},
 {"conv6aa",3,128,196},{"conv6a",3,196,196},{"conv6b",3,196,196},
 {"conv6_0",3,81,128},{"conv6_1",3,209,128},{"conv6_2",3,337,96},{"conv6_3",3,433,64},{"conv6_4",3,497,32},
 {"pf6",3,529,2},{"deconv6",4,2,2},{"upfeat6",4,529,2},
 {"conv5_0",3,213,128},{"conv5_1",3,341,128},{"conv5_2",3,469,96},{"conv5_3",3,565,64},{"conv5_4",3,629,32},
 {"pf5",3,661,2},{"deconv5",4,2,2},{"upfeat5",4,661,2},
 {"conv4_0",3,181,128},{"conv4_1",3,309,128},{"conv4_2",3,437,96},{"conv4_3",3,533,64},{"conv4_4",3,597,32},
 {"pf4",3,629,2},{"deconv4",4,2,2},{"upfeat4",4,629,2},
 {"conv3_0",3,149,128},{"conv3_1",3,277,128},{"conv3_2",3,405,96},{"conv3_3",3,501,64},{"conv3_4",3,565,32},
 {"pf3",3,597,2},{"deconv3",4,2,2},{"upfeat3",4,597,2},
 {"conv2_0",3,117,128},{"conv2_1",3,245,128},{"conv2_2",3,373,96},{"conv2_3",3,469,64},{"conv2_4",3,533,32},
 {"pf2",3,565,2},
 {"dc1",3,565,128},{"dc2",3,128,128},{"dc3",3,128,128},{"dc4",3,128,96},{"dc5",3,96,64},{"dc6",3,64,32},{"dc7",3,32,2}
};

static int li_(const char* n) {
  for (int i = 0; i < 63; ++i) if (!strcmp(SPECS[i].n, n)) return i;
  return 0;
}
static int pad32(int x) { return (x + 31) & ~31; }
static int pad16(int x) { return (x + 15) & ~15; }

extern "C" void kernel_launch(void* const* d_in, const int* in_sizes, int n_in,
                              void* d_out, int out_size, void* d_ws, size_t ws_size,
                              hipStream_t stream) {
  (void)in_sizes; (void)n_in; (void)out_size; (void)ws_size;
  const int N = 2;

  // ---------------- phase 1: pure pointer allocation (bump + slack) --------
  char* base = (char*)d_ws; size_t off = 0;
  auto af = [&](long n) -> f16* {
    off = (off + 255) & ~(size_t)255;
    f16* p = (f16*)(base + off);
    off += ((size_t)n + 1024) * sizeof(f16);   // slack for A-fragment over-read
    return p;
  };

  f16* wp[63]; int cinP[63], coutP[63];
  const float* wb[63];
  for (int i = 0; i < 63; ++i) {
    cinP[i]  = pad32(SPECS[i].cin);
    coutP[i] = pad16(SPECS[i].cout);
    wp[i] = af((long)SPECS[i].k * SPECS[i].k * cinP[i] * coutP[i]);
    wb[i] = (const float*)d_in[2 + 2 * i];
  }

  long npix0 = (long)N * 384 * 512;
  f16* im1 = af(npix0 * 8);
  f16* im2 = af(npix0 * 8);

  const int Hl[6] = {192, 96, 48, 24, 12, 6};
  const int Wl[6] = {256, 128, 64, 32, 16, 8};
  const int Cl[6]  = {16, 32, 64, 96, 128, 196};
  const int Clp[6] = {16, 32, 64, 96, 128, 200};   // padded strides (%8==0)
  f16 *P1[6], *P2[6];
  for (int l = 0; l < 6; ++l) {
    P1[l] = af((long)N * Hl[l] * Wl[l] * Clp[l]);
    P2[l] = af((long)N * Hl[l] * Wl[l] * Clp[l]);
  }
  f16* t0 = af((long)N * 192 * 256 * 16);
  f16* t1 = af((long)N * 192 * 256 * 16);

  const int cssL[5] = {529, 661, 629, 597, 565};   // logical widths L6..L2
  const int cssP[5] = {536, 664, 632, 600, 568};   // padded strides (%8==0)
  const int Hx[5]  = {6, 12, 24, 48, 96};
  const int Wx[5]  = {8, 16, 32, 64, 128};
  const int Cc[5]  = {196, 128, 96, 64, 32};
  const float scl[5] = {0.f, 0.625f, 1.25f, 2.5f, 5.0f};
  f16* X[5];
  for (int k = 0; k < 5; ++k) X[k] = af((long)N * Hx[k] * Wx[k] * cssP[k]);
  f16* flowb = af((long)N * 96 * 128 * 2);
  f16* warpb = af((long)N * 96 * 128 * 32);
  f16* dcA   = af((long)N * 96 * 128 * 128);
  f16* dcB   = af((long)N * 96 * 128 * 128);
  f16* dc7o  = af((long)N * 96 * 128 * 2);

  // ---------------- phase 2: zero whole workspace, pack weights ------------
  long n16 = (long)((off + 15) / 16);
  fill0_kernel<<<dim3((unsigned)((n16 + 255) / 256)), 256, 0, stream>>>((u32x4*)base, n16);

  for (int i = 0; i < 63; ++i) {
    int KK = SPECS[i].k * SPECS[i].k;
    long n = (long)KK * cinP[i] * coutP[i];
    pack_w_kernel<<<dim3((unsigned)((n + 255) / 256)), 256, 0, stream>>>(
        (const float*)d_in[1 + 2 * i], wp[i], KK, SPECS[i].cin, SPECS[i].cout,
        cinP[i], coutP[i]);
  }

  auto CONV = [&](const char* name, const f16* in, int in_cs, int in_coff,
                  f16* out, int out_cs, int out_coff,
                  int H, int W, int stride, int dil, int leaky) {
    int i = li_(name); const SpecT& s = SPECS[i];
    int OH = (H + stride - 1) / stride, OW = (W + stride - 1) / stride;
    int pth = (OH - 1) * stride + (s.k - 1) * dil + 1 - H; if (pth < 0) pth = 0;
    int ptw = (OW - 1) * stride + (s.k - 1) * dil + 1 - W; if (ptw < 0) ptw = 0;
    dim3 g((unsigned)((OH * OW + 127) / 128), (unsigned)((coutP[i] + 63) / 64), (unsigned)N);
    conv_wmma_kernel<<<g, 256, 0, stream>>>(in, in_cs, in_coff, cinP[i],
        out, out_cs, out_coff, s.cout, coutP[i], wp[i], wb[i],
        H, W, OH, OW, s.k, stride, dil, pth / 2, ptw / 2, leaky);
  };

  auto DECONV = [&](const char* name, const f16* in, int in_cs, int in_coff,
                    f16* out, int out_cs, int out_coff, int IH, int IW) {
    int i = li_(name); const SpecT& s = SPECS[i];
    long tot = (long)N * IH * 2 * IW * 2 * s.cout;
    deconv_kernel<<<dim3((unsigned)((tot + 255) / 256)), 256, 0, stream>>>(
        in, in_cs, in_coff, s.cin, wp[i], cinP[i], coutP[i], wb[i], s.cout,
        out, out_cs, out_coff, N, IH, IW, IH * 2, IW * 2);
  };

  // ---------------- phase 3: network -------------------------------------
  split_kernel<<<dim3((unsigned)((npix0 * 3 + 255) / 256)), 256, 0, stream>>>(
      (const float*)d_in[0], im1, im2, npix0);

  static const char* PYR[6][3] = {
      {"conv1a","conv1aa","conv1b"}, {"conv2a","conv2aa","conv2b"},
      {"conv3a","conv3aa","conv3b"}, {"conv4a","conv4aa","conv4b"},
      {"conv5a","conv5aa","conv5b"}, {"conv6aa","conv6a","conv6b"}};
  auto PYRAMID = [&](const f16* im, f16** P) {
    const f16* x = im; int xcs = 8; int h = 384, w = 512;
    for (int l = 0; l < 6; ++l) {
      CONV(PYR[l][0], x, xcs, 0, t0, Clp[l], 0, h, w, 2, 1, 1);
      h = Hl[l]; w = Wl[l];
      CONV(PYR[l][1], t0, Clp[l], 0, t1, Clp[l], 0, h, w, 1, 1, 1);
      CONV(PYR[l][2], t1, Clp[l], 0, P[l], Clp[l], 0, h, w, 1, 1, 1);
      x = P[l]; xcs = Clp[l];
    }
  };
  PYRAMID(im1, P1);
  PYRAMID(im2, P2);

  auto DENSE = [&](int lvl, f16* Xb, int cs, int H, int W) {
    static const int ico[5] = {448, 320, 192, 96, 32};
    static const int oco[5] = {320, 192, 96, 32, 0};
    char nm[16];
    for (int i = 0; i < 5; ++i) {
      snprintf(nm, sizeof nm, "conv%d_%d", lvl, i);
      CONV(nm, Xb, cs, ico[i], Xb, cs, oco[i], H, W, 1, 1, 1);
    }
  };

  // ---- level 6 ----
  {
    long tcv = (long)N * 6 * 8 * 81;
    cost_volume_kernel<<<dim3((unsigned)((tcv + 255) / 256)), 256, 0, stream>>>(
        P1[5], Clp[5], P2[5], Clp[5], 196, X[0], cssP[0], 448, N, 6, 8);
    DENSE(6, X[0], cssP[0], 6, 8);
    CONV("pf6", X[0], cssP[0], 0, flowb, 2, 0, 6, 8, 1, 1, 0);
    DECONV("deconv6", flowb, 2, 0, X[1], cssP[1], cssL[1] - 4, 6, 8);   // up_flow6
    DECONV("upfeat6", X[0], cssP[0], 0, X[1], cssP[1], cssL[1] - 2, 6, 8);
  }

  // ---- levels 5..2 ----
  for (int k = 1; k < 5; ++k) {
    int lvl = 6 - k;
    int H = Hx[k], W = Wx[k], C = Cc[k];
    int csP = cssP[k], csL = cssL[k];
    f16* Xb = X[k];
    const f16* c1 = P1[5 - k];
    const f16* c2 = P2[5 - k];
    int pcs = Clp[5 - k];
    long np = (long)N * H * W;

    warp_kernel<<<dim3((unsigned)((np * C + 255) / 256)), 256, 0, stream>>>(
        c2, pcs, C, Xb, csP, csL - 4, scl[k], warpb, N, H, W);
    cost_volume_kernel<<<dim3((unsigned)((np * 81 + 255) / 256)), 256, 0, stream>>>(
        c1, pcs, warpb, C, C, Xb, csP, 448, N, H, W);
    copy_ch_kernel<<<dim3((unsigned)((np * C + 255) / 256)), 256, 0, stream>>>(
        c1, pcs, 0, Xb, csP, 529, C, np);
    DENSE(lvl, Xb, csP, H, W);

    char nm[16];
    snprintf(nm, sizeof nm, "pf%d", lvl);
    CONV(nm, Xb, csP, 0, flowb, 2, 0, H, W, 1, 1, 0);
    if (lvl > 2) {
      snprintf(nm, sizeof nm, "deconv%d", lvl);
      DECONV(nm, flowb, 2, 0, X[k + 1], cssP[k + 1], cssL[k + 1] - 4, H, W);
      snprintf(nm, sizeof nm, "upfeat%d", lvl);
      DECONV(nm, Xb, csP, 0, X[k + 1], cssP[k + 1], cssL[k + 1] - 2, H, W);
    }
  }

  // ---- dilated context network at level 2 ----
  CONV("dc1", X[4], cssP[4], 0, dcA, 128, 0, 96, 128, 1, 1, 1);
  CONV("dc2", dcA, 128, 0, dcB, 128, 0, 96, 128, 1, 2, 1);
  CONV("dc3", dcB, 128, 0, dcA, 128, 0, 96, 128, 1, 4, 1);
  CONV("dc4", dcA, 128, 0, dcB, 96, 0, 96, 128, 1, 8, 1);
  CONV("dc5", dcB, 96, 0, dcA, 64, 0, 96, 128, 1, 16, 1);
  CONV("dc6", dcA, 64, 0, dcB, 32, 0, 96, 128, 1, 1, 1);
  CONV("dc7", dcB, 32, 0, dc7o, 2, 0, 96, 128, 1, 1, 0);

  long nf = (long)N * 96 * 128 * 2;
  add_flow_kernel<<<dim3((unsigned)((nf + 255) / 256)), 256, 0, stream>>>(
      flowb, dc7o, (float*)d_out, nf);
}
